// WindowAttentionSparse_22213570855201
// MI455X (gfx1250) — matrix operand order repaired and checked by reading, not compiled, not run on hardware
//
#include <hip/hip_runtime.h>
#include <hip/hip_bf16.h>
#include <stdint.h>

typedef __attribute__((ext_vector_type(16))) _Float16 v16h;
typedef __attribute__((ext_vector_type(8)))  float    v8f;

// Problem constants
constexpr int S   = 64;    // window volume
constexpr int DIM = 192;   // channels
constexpr int H   = 6;     // heads
constexpr int Ch  = 32;    // head dim
constexpr int NWIN = 2048;

// Packed-weight fragment geometry (B operand of v_wmma_f32_16x16x32_f16)
constexpr int KT_QKV  = DIM / 32;      // 6 K-tiles
constexpr int NT_QKV  = 3 * DIM / 16;  // 36 N-tiles (576 cols)
constexpr int NT_PROJ = DIM / 16;      // 12 N-tiles
constexpr int FRAG_H  = 512;           // halfs per 32x16 fragment (32 lanes * 16)
constexpr int NQKV  = KT_QKV * NT_QKV  * FRAG_H;  // 110592 halfs
constexpr int NPROJ = KT_QKV * NT_PROJ * FRAG_H;  //  36864 halfs

// LDS strides (elements), padded against bank conflicts
constexpr int XS_STRIDE = 200;  // halfs  (192 + 8)
constexpr int QK_STRIDE = 36;   // halfs  (32 + 4)
constexpr int VT_STRIDE = 68;   // halfs  (64 + 4)
constexpr int SC_STRIDE = 66;   // floats (64 + 2)

// LDS layout (bytes)
constexpr int OFF_XS      = 0;                                   // x tile / attn-out tile (f16)
constexpr int OFF_QS      = OFF_XS + S * XS_STRIDE * 2;          // q (pre-scaled, f16)
constexpr int OFF_KS      = OFF_QS + H * S * QK_STRIDE * 2;      // k (f16)
constexpr int OFF_VT      = OFF_KS + H * S * QK_STRIDE * 2;      // v transposed (f16)
constexpr int OFF_SC      = OFF_VT + H * Ch * VT_STRIDE * 2;     // scores (f32), one head
constexpr int OFF_MASKADD = OFF_SC + S * SC_STRIDE * 4;          // column mask add (f32[64])
constexpr int OFF_VALID01 = OFF_MASKADD + S * 4;                 // row multiplier (f32[64])
constexpr int OFF_REGION  = OFF_VALID01 + S * 4;                 // region ids (int[64])
constexpr int OFF_BQ      = OFF_REGION + S * 4;                  // b_qkv (f32[576])
constexpr int OFF_BP      = OFF_BQ + 3 * DIM * 4;                // b_proj (f32[192])
constexpr int SMEM_BYTES  = OFF_BP + DIM * 4;                    // = 127744
static_assert(SMEM_BYTES == 127744, "LDS layout changed");

// ---------------- fragment loaders (CDNA5 wave32 WMMA layouts) ----------------

// A (16x32 f16): lane L -> row M=L%16; lanes 0-15 hold K 0..15,
// lanes 16-31 hold K 16..31; VGPR g packs K pair (2g, 2g+1).
__device__ __forceinline__ v16h load_fragA_f16(const _Float16* base, int row0,
                                               int col0, int stride, int lane) {
  union { v16h v; uint32_t u[8]; } f;
  const int r = row0 + (lane & 15);
  const int hi8 = ((lane >> 4) & 1) * 16;
  const _Float16* p = base + r * stride + col0 + hi8;
#pragma unroll
  for (int g = 0; g < 8; ++g) {
    const int koff = 2 * g;
    f.u[g] = *(const uint32_t*)(p + koff);
  }
  return f.v;
}

// Same A layout but sourced from an f32 array (softmaxed scores) with convert.
__device__ __forceinline__ v16h load_fragA_f32(const float* base, int row0,
                                               int col0, int stride, int lane) {
  union { v16h v; _Float16 h[16]; } f;
  const int r = row0 + (lane & 15);
  const int hi8 = ((lane >> 4) & 1) * 16;
  const float* p = base + r * stride + col0 + hi8;
#pragma unroll
  for (int g = 0; g < 8; ++g) {
    const int koff = 2 * g;
    float2 t = *(const float2*)(p + koff);
    f.h[2 * g]     = (_Float16)t.x;
    f.h[2 * g + 1] = (_Float16)t.y;
  }
  return f.v;
}

// B (32x16 f16): lane L -> col N=L%16; lanes 0-15 hold K=0..15, lanes 16-31
// K=16..31; VGPR g packs K pair (2g, 2g+1).  `base` is stored N-major
// (row = N, inner = K) so pairs are contiguous.
__device__ __forceinline__ v16h load_fragB_f16(const _Float16* base, int row0,
                                               int col0, int stride, int lane) {
  union { v16h v; uint32_t u[8]; } f;
  const int r = row0 + (lane & 15);
  const int kb = col0 + ((lane >> 4) & 1) * 16;
  const _Float16* p = base + r * stride + kb;
#pragma unroll
  for (int g = 0; g < 8; ++g) f.u[g] = *(const uint32_t*)(p + 2 * g);
  return f.v;
}

// Pre-swizzled B fragment from workspace.  `fb` points at this tile's first
// fragment, already offset by lane*16 halfs; kt selects the K-step via an
// immediate offset (kt * 1024 bytes), so all 12 b128 loads of a tile share
// one 64-bit base address.
__device__ __forceinline__ v16h load_fragB_pk(const uint4* fb, int kt) {
  union { v16h v; uint4 q[2]; } f;
  f.q[0] = fb[kt * 64];
  f.q[1] = fb[kt * 64 + 1];
  return f.v;
}

// ---------------- prep kernel: pack f32 weights into f16 B fragments ----------------

__global__ __launch_bounds__(256) void pack_weights_kernel(
    const float* __restrict__ w_qkv, const float* __restrict__ w_proj,
    _Float16* __restrict__ ws) {
  int idx = blockIdx.x * blockDim.x + threadIdx.x;
  if (idx >= NQKV + NPROJ) return;
  const float* src;
  int cols, j = idx;
  if (j < NQKV) { src = w_qkv; cols = 3 * DIM; }
  else          { src = w_proj; cols = DIM; j -= NQKV; }
  const int frag   = j >> 9;        // 512 halfs per fragment
  const int within = j & 511;
  const int lane   = within >> 4;
  const int hh     = within & 15;
  const int nt = frag / KT_QKV;
  const int kt = frag % KT_QKV;
  const int g = hh >> 1, p = hh & 1;
  const int n = nt * 16 + (lane & 15);
  const int k = kt * 32 + ((lane >> 4) << 4) + 2 * g + p;  // B layout
  ws[idx] = (_Float16)src[k * cols + n];
}

// ---------------- main kernel: one workgroup per window ----------------

__global__ __launch_bounds__(256) void win_attn_kernel(
    const float* __restrict__ x, const int* __restrict__ valid,
    const int* __restrict__ region, const float* __restrict__ b_qkv,
    const float* __restrict__ rpb, const float* __restrict__ b_proj,
    const _Float16* __restrict__ wpk, float* __restrict__ out) {
  extern __shared__ char smem[];
  _Float16* xs      = (_Float16*)(smem + OFF_XS);
  _Float16* qs      = (_Float16*)(smem + OFF_QS);
  _Float16* ks      = (_Float16*)(smem + OFF_KS);
  _Float16* vT      = (_Float16*)(smem + OFF_VT);
  float*    sc      = (float*)(smem + OFF_SC);
  float*    maskadd = (float*)(smem + OFF_MASKADD);
  float*    valid01 = (float*)(smem + OFF_VALID01);
  int*      sregion = (int*)(smem + OFF_REGION);
  float*    sbq     = (float*)(smem + OFF_BQ);
  float*    sbp     = (float*)(smem + OFF_BP);

  const int t = threadIdx.x;
  const int lane = t & 31;
  // Wave id is uniform across the wave: force it into an SGPR so every tile
  // coordinate / section branch below becomes scalar control flow, not
  // EXEC-mask divergence around the WMMA stores.
  const int wv = __builtin_amdgcn_readfirstlane(t >> 5);   // 8 wave32 per block
  const int b = blockIdx.x;
  const int nl = lane & 15;
  const int hi8 = ((lane >> 4) & 1) * 8;

  // Phase 0: per-window metadata + bias preload
  if (t < S) {
    const int v = valid[b * S + t];
    maskadd[t] = v ? 0.0f : -1e9f;
    valid01[t] = v ? 1.0f : 0.0f;
    sregion[t] = region[b * S + t];
  }
  for (int i = t; i < 3 * DIM; i += 256) sbq[i] = b_qkv[i];
  if (t < DIM) sbp[t] = b_proj[t];

  // Phase 1: x[b] (64x192 f32) -> xs (f16)
  {
    const float4* xb = (const float4*)(x + (size_t)b * S * DIM);
    for (int i = t; i < S * DIM / 4; i += 256) {
      float4 v = xb[i];
      const int e = i * 4, s = e / DIM, d = e % DIM;
      _Float16* p = xs + s * XS_STRIDE + d;
      p[0] = (_Float16)v.x; p[1] = (_Float16)v.y;
      p[2] = (_Float16)v.z; p[3] = (_Float16)v.w;
    }
  }
  __syncthreads();

  // Phase 2: QKV GEMM  [64,192] x [192,576], 144 tiles over 8 waves.
  // mt = wv&3 is wave-invariant -> the 6 A fragments stay resident in VGPRs
  // across the whole loop; only B fragments stream from L2.
  for (int i = 0; i < 18; ++i) {
    const int tau = i * 8 + wv;              // uniform
    const int mt = tau & 3, nt = tau >> 2;   // uniform
    const uint4* fb =
        (const uint4*)(wpk + (size_t)nt * KT_QKV * FRAG_H) + lane * 2;
    v8f acc = {0.f, 0.f, 0.f, 0.f, 0.f, 0.f, 0.f, 0.f};
#pragma unroll
    for (int kt = 0; kt < KT_QKV; ++kt) {
      v16h a  = load_fragA_f16(xs, mt * 16, kt * 32, XS_STRIDE, lane);
      v16h bm = load_fragB_pk(fb, kt);
      acc = __builtin_amdgcn_wmma_f32_16x16x32_f16(false, a, false, bm,
                                                   (short)0, acc, false, false);
    }
    const int sec = nt / 12;             // 0=q 1=k 2=v (uniform)
    const int hd  = (nt % 12) >> 1;      // head (uniform)
    const int cb  = (nt & 1) * 16;       // channel base within head (uniform)
    const float bias = sbq[nt * 16 + nl];
    if (sec == 0) {
      _Float16* dst = qs + hd * S * QK_STRIDE + (mt * 16 + hi8) * QK_STRIDE + cb + nl;
#pragma unroll
      for (int r = 0; r < 8; ++r)
        dst[r * QK_STRIDE] = (_Float16)((acc[r] + bias) * 0.17677669529663687f);
    } else if (sec == 1) {
      _Float16* dst = ks + hd * S * QK_STRIDE + (mt * 16 + hi8) * QK_STRIDE + cb + nl;
#pragma unroll
      for (int r = 0; r < 8; ++r)
        dst[r * QK_STRIDE] = (_Float16)(acc[r] + bias);
    } else {
      _Float16* dst = vT + hd * Ch * VT_STRIDE + (cb + nl) * VT_STRIDE + mt * 16 + hi8;
#pragma unroll
      for (int r = 0; r < 8; ++r)
        dst[r] = (_Float16)(acc[r] + bias);
    }
  }
  __syncthreads();

  // Phase 3: attention, head by head (scores LDS reused)
  for (int h = 0; h < H; ++h) {
    const _Float16* qh = qs + h * S * QK_STRIDE;
    const _Float16* kh = ks + h * S * QK_STRIDE;
    const _Float16* vh = vT + h * Ch * VT_STRIDE;

    // 3a: scores = q k^T + rpb + cross-region + valid-column mask
    for (int i = 0; i < 2; ++i) {
      const int tau = wv * 2 + i;              // uniform
      const int it = tau >> 2, jt = tau & 3;   // uniform
      v16h a  = load_fragA_f16(qh, it * 16, 0, QK_STRIDE, lane);
      v16h bm = load_fragB_f16(kh, jt * 16, 0, QK_STRIDE, lane);  // k rows = cols of k^T
      v8f acc = {0.f, 0.f, 0.f, 0.f, 0.f, 0.f, 0.f, 0.f};
      acc = __builtin_amdgcn_wmma_f32_16x16x32_f16(false, a, false, bm,
                                                   (short)0, acc, false, false);
      const int jj = jt * 16 + nl;
      const int rj = sregion[jj];
      const float madd = maskadd[jj];
#pragma unroll
      for (int r = 0; r < 8; ++r) {
        const int ii = it * 16 + r + hi8;
        float v = acc[r] + rpb[((h * S) + ii) * S + jj];
        v += (sregion[ii] != rj) ? -100.0f : 0.0f;
        v += madd;
        sc[ii * SC_STRIDE + jj] = v;
      }
    }
    __syncthreads();

    // 3b: row softmax (8 rows per wave, 2 cols per lane, wave32 reductions)
    for (int ir = 0; ir < 8; ++ir) {
      const int row = wv * 8 + ir;
      float v0 = sc[row * SC_STRIDE + lane];
      float v1 = sc[row * SC_STRIDE + lane + 32];
      float m = fmaxf(v0, v1);
      for (int off = 16; off; off >>= 1) m = fmaxf(m, __shfl_xor(m, off, 32));
      const float e0 = __expf(v0 - m);
      const float e1 = __expf(v1 - m);
      float ssum = e0 + e1;
      for (int off = 16; off; off >>= 1) ssum += __shfl_xor(ssum, off, 32);
      const float inv = 1.0f / ssum;
      sc[row * SC_STRIDE + lane]      = e0 * inv;
      sc[row * SC_STRIDE + lane + 32] = e1 * inv;
    }
    __syncthreads();

    // 3c: out_h = attn @ v   (8 tiles = 1 per wave, K = 64)
    {
      const int it = wv >> 1, ct = wv & 1;     // uniform
      v8f acc = {0.f, 0.f, 0.f, 0.f, 0.f, 0.f, 0.f, 0.f};
#pragma unroll
      for (int kt = 0; kt < 2; ++kt) {
        v16h a  = load_fragA_f32(sc, it * 16, kt * 32, SC_STRIDE, lane);
        v16h bm = load_fragB_f16(vh, ct * 16, kt * 32, VT_STRIDE, lane);  // vT rows = N
        acc = __builtin_amdgcn_wmma_f32_16x16x32_f16(false, a, false, bm,
                                                     (short)0, acc, false, false);
      }
      _Float16* dst = xs + (it * 16 + hi8) * XS_STRIDE + h * Ch + ct * 16 + nl;
#pragma unroll
      for (int r = 0; r < 8; ++r) dst[r * XS_STRIDE] = (_Float16)acc[r];
    }
    __syncthreads();
  }

  // Phase 4: projection  [64,192] x [192,192] + bias, row valid mask, store
  {
    const _Float16* wp = wpk + NQKV;
    for (int i = 0; i < 6; ++i) {
      const int tau = i * 8 + wv;              // uniform
      const int mt = tau & 3, nt = tau >> 2;   // uniform
      const uint4* fb =
          (const uint4*)(wp + (size_t)nt * KT_QKV * FRAG_H) + lane * 2;
      v8f acc = {0.f, 0.f, 0.f, 0.f, 0.f, 0.f, 0.f, 0.f};
#pragma unroll
      for (int kt = 0; kt < KT_QKV; ++kt) {
        v16h a  = load_fragA_f16(xs, mt * 16, kt * 32, XS_STRIDE, lane);
        v16h bm = load_fragB_pk(fb, kt);
        acc = __builtin_amdgcn_wmma_f32_16x16x32_f16(false, a, false, bm,
                                                     (short)0, acc, false, false);
      }
      const float bias = sbp[nt * 16 + nl];
      float* dst = out + ((size_t)b * S + mt * 16 + hi8) * DIM + nt * 16 + nl;
#pragma unroll
      for (int r = 0; r < 8; ++r)
        dst[r * DIM] = (acc[r] + bias) * valid01[mt * 16 + r + hi8];
    }
  }
}

// ---------------- host entry ----------------

extern "C" void kernel_launch(void* const* d_in, const int* in_sizes, int n_in,
                              void* d_out, int out_size, void* d_ws, size_t ws_size,
                              hipStream_t stream) {
  (void)in_sizes; (void)n_in; (void)out_size; (void)ws_size;
  const float* x      = (const float*)d_in[0];
  const int*   valid  = (const int*)d_in[1];
  const int*   region = (const int*)d_in[2];
  const float* w_qkv  = (const float*)d_in[3];
  const float* b_qkv  = (const float*)d_in[4];
  const float* rpb    = (const float*)d_in[5];
  const float* w_proj = (const float*)d_in[6];
  const float* b_proj = (const float*)d_in[7];
  _Float16* wpk = (_Float16*)d_ws;   // 294912 bytes of packed f16 weights
  float* out = (float*)d_out;

  const int packN = NQKV + NPROJ;
  pack_weights_kernel<<<(packN + 255) / 256, 256, 0, stream>>>(w_qkv, w_proj, wpk);

  hipFuncSetAttribute(reinterpret_cast<const void*>(win_attn_kernel),
                      hipFuncAttributeMaxDynamicSharedMemorySize, SMEM_BYTES);
  win_attn_kernel<<<NWIN, 256, SMEM_BYTES, stream>>>(x, valid, region, b_qkv,
                                                     rpb, b_proj, wpk, out);
}
